// SparseMoEBlock_30356828848188
// MI455X (gfx1250) — compile-verified, matrix operand order checked by reading, not run on hardware
//
#include <hip/hip_runtime.h>
#include <hip/hip_bf16.h>
#include <math.h>

typedef __attribute__((ext_vector_type(16))) __bf16 v16bf;
typedef __attribute__((ext_vector_type(8)))  float  v8f;
typedef unsigned short ushort_t;
typedef unsigned int   uint_t;

#define T_TOK 4096
#define D_DIM 1024
#define H_DIM 2048
#define E_NUM 8
#define TOPK  2

// ---------- helpers ----------

__device__ inline ushort_t f2bf(float f) {
    uint_t x = __builtin_bit_cast(uint_t, f);
    uint_t r = (x + 0x7FFFu + ((x >> 16) & 1u)) >> 16;   // round-to-nearest-even
    return (ushort_t)r;
}

// branch-free tanh-approx gelu using native v_exp_f32
__device__ inline float gelu_tanh(float v) {
    const float c = 0.7978845608028654f;
    float u = c * (v + 0.044715f * v * v * v);
    float t = 1.0f - 2.0f / (1.0f + __expf(2.0f * u));   // tanh(u)
    return 0.5f * v * (1.0f + t);
}

// Packed fragment layouts: each 32Kx16N tile is 32 lanes x 8 dwords (1 KB);
// a lane's 8 dwords are contiguous -> fragment = two 16-byte vector loads.

// A fragment (16x32 bf16) from LDS packed [kt][lane][8 dwords]
__device__ inline v16bf load_a_packed(const uint_t* s, int kt, int lane) {
    union { v16bf v; uint4 q[2]; } a;
    const uint4* p = (const uint4*)(s + kt * 256 + lane * 8);
    a.q[0] = p[0];
    a.q[1] = p[1];
    return a.v;
}

// B fragment (32x16 bf16) from global packed [nt][kt][lane][8 dwords]
__device__ inline v16bf load_b_packed(const uint_t* w, int KT, int nt, int kt, int lane) {
    union { v16bf v; uint4 q[2]; } b;
    const uint4* p = (const uint4*)(w + (((size_t)nt * KT + kt) * 32 + lane) * 8);
    b.q[0] = p[0];
    b.q[1] = p[1];
    return b.v;
}

// Store one C column (8 rows) into LDS packed-A layout (for GEMM2's A input).
__device__ inline void store_h_packed(ushort_t* hs, int kdim, int g, v8f acc, float bias) {
    int kt   = kdim >> 5;
    int kk   = kdim & 31;
    int half = kk & 1;
    int q    = kk >> 1;
    int gd   = (q >> 2) & 1;
    int vd   = (q & 3) | ((q >> 3) << 2);
    int base = (kt * 256 + (16 * gd) * 8 + vd) * 2 + half;
#pragma unroll
    for (int r = 0; r < 8; ++r) {
        int m = r + 8 * g;
        hs[base + m * 16] = f2bf(gelu_tanh(acc[r] + bias));
    }
}

// ---------- kernels ----------

__global__ void moe_init(float* out, int n, int* counts, float* psum) {
    int stride = gridDim.x * blockDim.x;
    for (int i = blockIdx.x * blockDim.x + threadIdx.x; i < n; i += stride) out[i] = 0.0f;
    if (blockIdx.x == 0 && threadIdx.x < E_NUM) {
        counts[threadIdx.x] = 0;
        psum[threadIdx.x]   = 0.0f;
    }
}

// fp32 [E][K][N] -> bf16 packed WMMA-B tiles [e][nt][kt][lane][8 dwords]
// per-expert matrix = 2^20 dwords for both W1 (KT=32,NT=128) and W2 (KT=64,NT=64)
__global__ void moe_convert_pack(const float* __restrict__ src, uint_t* __restrict__ dst,
                                 int kt_shift, int K_total, int N_total) {
    const int perE = 1 << 20;
    int stride = gridDim.x * blockDim.x;
    for (int o = blockIdx.x * blockDim.x + threadIdx.x; o < E_NUM * perE; o += stride) {
        int e     = o >> 20;
        int rem   = o & (perE - 1);
        int v     = rem & 7;
        int lane  = (rem >> 3) & 31;
        int tileI = rem >> 8;
        int kt    = tileI & ((1 << kt_shift) - 1);
        int nt    = tileI >> kt_shift;
        int n     = nt * 16 + (lane & 15);
        int g     = lane >> 4;
        int kk    = ((v & 4) ? 16 : 0) + 8 * g + 2 * (v & 3);
        const float* s = src + (size_t)e * K_total * N_total
                             + (size_t)(kt * 32 + kk) * N_total + n;
        uint_t lo = f2bf(s[0]);
        uint_t hi = f2bf(s[N_total]);
        dst[o] = lo | (hi << 16);
    }
}

// one block (256 thr = 8 waves) per token; wave e computes logit for expert e
__global__ void moe_gate(const float* __restrict__ x, const float* __restrict__ Wr,
                         const float* __restrict__ br,
                         int* counts, float* psum, int* tokL, float* wL) {
    int t    = blockIdx.x;
    int e    = threadIdx.x >> 5;
    int lane = threadIdx.x & 31;
    const float* xr = x + (size_t)t * D_DIM;

    float s = 0.0f;
    for (int d = lane; d < D_DIM; d += 32) s += xr[d] * Wr[d * E_NUM + e];
#pragma unroll
    for (int off = 16; off > 0; off >>= 1) s += __shfl_down(s, off, 32);

    __shared__ float logits[E_NUM];
    if (lane == 0) logits[e] = s + br[e];
    __syncthreads();

    if (threadIdx.x == 0) {
        float l[E_NUM], p[E_NUM];
        float mx = -1e30f;
#pragma unroll
        for (int i = 0; i < E_NUM; ++i) { l[i] = logits[i]; mx = fmaxf(mx, l[i]); }
        float se = 0.0f;
#pragma unroll
        for (int i = 0; i < E_NUM; ++i) { p[i] = __expf(l[i] - mx); se += p[i]; }
        float inv = 1.0f / se;
#pragma unroll
        for (int i = 0; i < E_NUM; ++i) atomicAdd(&psum[i], p[i] * inv);

        int i0 = 0;
#pragma unroll
        for (int i = 1; i < E_NUM; ++i) if (l[i] > l[i0]) i0 = i;
        int i1 = (i0 == 0) ? 1 : 0;
#pragma unroll
        for (int i = 0; i < E_NUM; ++i) if (i != i0 && l[i] > l[i1]) i1 = i;

        float r0 = 1.0f / (1.0f + __expf(l[i1] - l[i0]));
        float r1 = 1.0f - r0;

        int s0 = atomicAdd(&counts[i0], 1);
        tokL[i0 * T_TOK + s0] = t;  wL[i0 * T_TOK + s0] = r0;
        int s1 = atomicAdd(&counts[i1], 1);
        tokL[i1 * T_TOK + s1] = t;  wL[i1 * T_TOK + s1] = r1;
    }
}

__global__ void moe_lbal(const int* counts, const float* psum, float* out_lbal) {
    if (threadIdx.x == 0 && blockIdx.x == 0) {
        float acc = 0.0f;
#pragma unroll
        for (int e = 0; e < E_NUM; ++e) {
            float f = (float)counts[e] / (float)(T_TOK * TOPK);
            float P = psum[e] / (float)T_TOK;
            acc += f * P;
        }
        *out_lbal = (float)E_NUM * acc;
    }
}

// blockIdx.x encodes (tile, expert); each block = 16 tokens of one expert, 8 waves.
// Each wave processes 4 N-tiles per A-fragment load: per k-step
// 2x ds_load_b128 + 8x global_load_b128 + 4x v_wmma (2.5 loads per wmma).
__global__ void moe_experts(const float* __restrict__ x,
                            const uint_t* __restrict__ W1p, const float* __restrict__ b1,
                            const uint_t* __restrict__ W2p, const float* __restrict__ b2,
                            const int* __restrict__ counts,
                            const int* __restrict__ tokL, const float* __restrict__ wL,
                            float* __restrict__ out) {
    extern __shared__ __align__(16) uint_t smem[];
    uint_t*   xsu  = smem;           // x tile, packed A: 32 kt-tiles * 256 dw = 32 KB
    uint_t*   hsu  = smem + 8192;    // h tile, packed A: 64 kt-tiles * 256 dw = 64 KB
    ushort_t* hs16 = (ushort_t*)hsu;
    __shared__ int   toks[16];
    __shared__ float wgts[16];

    const int e    = blockIdx.x & (E_NUM - 1);
    const int tile = blockIdx.x >> 3;
    const int cnt  = counts[e];
    if (tile * 16 >= cnt) return;

    const int tid  = threadIdx.x;
    const int wave = tid >> 5;
    const int lane = tid & 31;
    const int g    = lane >> 4;
    const int nc   = lane & 15;

    if (tid < 16) {
        int idx = tile * 16 + tid;
        if (idx < cnt) { toks[tid] = tokL[e * T_TOK + idx]; wgts[tid] = wL[e * T_TOK + idx]; }
        else           { toks[tid] = 0;                     wgts[tid] = 0.0f; }
    }
    __syncthreads();

    // stage x tile -> LDS bf16 in packed-A layout (coalesced 8B global reads)
    for (int i = tid; i < 16 * (D_DIM / 2); i += 256) {
        int m  = i >> 9;            // token row 0..15
        int kp = i & 511;           // K pair index
        const float2 xv = *(const float2*)(x + (size_t)toks[m] * D_DIM + 2 * kp);
        int kt = kp >> 4;
        int q  = kp & 15;
        int gg = (q >> 2) & 1;
        int vv = (q & 3) | ((q >> 3) << 2);
        xsu[kt * 256 + (m + 16 * gg) * 8 + vv] =
            (uint_t)f2bf(xv.x) | ((uint_t)f2bf(xv.y) << 16);
    }
    __syncthreads();

    const int KT1 = D_DIM / 32;   // 32
    const int KT2 = H_DIM / 32;   // 64
    const uint_t* w1e = W1p + (size_t)e * (1 << 20);
    const uint_t* w2e = W2p + (size_t)e * (1 << 20);

    // GEMM1: h[16][H] = gelu(xs @ W1[e] + b1[e]); 128 N-tiles as 32 quads over 8 waves
    for (int p = wave; p < (H_DIM / 64); p += 8) {
        const int ntb = 4 * p;
        v8f acc0 = {}, acc1 = {}, acc2 = {}, acc3 = {};
#pragma unroll 4
        for (int kt = 0; kt < KT1; ++kt) {
            v16bf a  = load_a_packed(xsu, kt, lane);
            v16bf b0 = load_b_packed(w1e, KT1, ntb + 0, kt, lane);
            v16bf bb1 = load_b_packed(w1e, KT1, ntb + 1, kt, lane);
            v16bf bb2 = load_b_packed(w1e, KT1, ntb + 2, kt, lane);
            v16bf bb3 = load_b_packed(w1e, KT1, ntb + 3, kt, lane);
            acc0 = __builtin_amdgcn_wmma_f32_16x16x32_bf16(false, a, false, b0,
                                                           (short)0, acc0, false, false);
            acc1 = __builtin_amdgcn_wmma_f32_16x16x32_bf16(false, a, false, bb1,
                                                           (short)0, acc1, false, false);
            acc2 = __builtin_amdgcn_wmma_f32_16x16x32_bf16(false, a, false, bb2,
                                                           (short)0, acc2, false, false);
            acc3 = __builtin_amdgcn_wmma_f32_16x16x32_bf16(false, a, false, bb3,
                                                           (short)0, acc3, false, false);
        }
        int n0 = (ntb + 0) * 16 + nc;
        int n1 = (ntb + 1) * 16 + nc;
        int n2 = (ntb + 2) * 16 + nc;
        int n3 = (ntb + 3) * 16 + nc;
        store_h_packed(hs16, n0, g, acc0, b1[e * H_DIM + n0]);
        store_h_packed(hs16, n1, g, acc1, b1[e * H_DIM + n1]);
        store_h_packed(hs16, n2, g, acc2, b1[e * H_DIM + n2]);
        store_h_packed(hs16, n3, g, acc3, b1[e * H_DIM + n3]);
    }
    __syncthreads();

    // GEMM2: y[16][D] = hs @ W2[e] + b2[e]; out[tok] += w*y; 64 N-tiles as 16 quads
    for (int p = wave; p < (D_DIM / 64); p += 8) {
        const int ntb = 4 * p;
        v8f acc0 = {}, acc1 = {}, acc2 = {}, acc3 = {};
#pragma unroll 4
        for (int kt = 0; kt < KT2; ++kt) {
            v16bf a  = load_a_packed(hsu, kt, lane);
            v16bf b0 = load_b_packed(w2e, KT2, ntb + 0, kt, lane);
            v16bf bb1 = load_b_packed(w2e, KT2, ntb + 1, kt, lane);
            v16bf bb2 = load_b_packed(w2e, KT2, ntb + 2, kt, lane);
            v16bf bb3 = load_b_packed(w2e, KT2, ntb + 3, kt, lane);
            acc0 = __builtin_amdgcn_wmma_f32_16x16x32_bf16(false, a, false, b0,
                                                           (short)0, acc0, false, false);
            acc1 = __builtin_amdgcn_wmma_f32_16x16x32_bf16(false, a, false, bb1,
                                                           (short)0, acc1, false, false);
            acc2 = __builtin_amdgcn_wmma_f32_16x16x32_bf16(false, a, false, bb2,
                                                           (short)0, acc2, false, false);
            acc3 = __builtin_amdgcn_wmma_f32_16x16x32_bf16(false, a, false, bb3,
                                                           (short)0, acc3, false, false);
        }
        int n0 = (ntb + 0) * 16 + nc;
        int n1 = (ntb + 1) * 16 + nc;
        int n2 = (ntb + 2) * 16 + nc;
        int n3 = (ntb + 3) * 16 + nc;
        float bias0 = b2[e * D_DIM + n0];
        float bias1 = b2[e * D_DIM + n1];
        float bias2 = b2[e * D_DIM + n2];
        float bias3 = b2[e * D_DIM + n3];
#pragma unroll
        for (int r = 0; r < 8; ++r) {
            int m = r + 8 * g;
            float w = wgts[m];
            size_t row = (size_t)toks[m] * D_DIM;
            atomicAdd(&out[row + n0], (acc0[r] + bias0) * w);
            atomicAdd(&out[row + n1], (acc1[r] + bias1) * w);
            atomicAdd(&out[row + n2], (acc2[r] + bias2) * w);
            atomicAdd(&out[row + n3], (acc3[r] + bias3) * w);
        }
    }
}

// ---------- launch ----------

extern "C" void kernel_launch(void* const* d_in, const int* in_sizes, int n_in,
                              void* d_out, int out_size, void* d_ws, size_t ws_size,
                              hipStream_t stream) {
    const float* x  = (const float*)d_in[0];
    const float* Wr = (const float*)d_in[1];
    const float* br = (const float*)d_in[2];
    const float* W1 = (const float*)d_in[3];
    const float* b1 = (const float*)d_in[4];
    const float* W2 = (const float*)d_in[5];
    const float* b2 = (const float*)d_in[6];
    float* out = (float*)d_out;

    const size_t nW = (size_t)E_NUM * D_DIM * H_DIM;       // 16,777,216 elements

    // workspace layout
    char* ws = (char*)d_ws;
    uint_t* W1p  = (uint_t*)(ws);                          // 32 MB packed bf16
    uint_t* W2p  = (uint_t*)(ws + 2 * nW);                 // 32 MB packed bf16
    int*    cnts = (int*)   (ws + 4 * nW);
    float*  psum = (float*) (ws + 4 * nW + 128);
    int*    tokL = (int*)   (ws + 4 * nW + 256);
    float*  wLst = (float*) (ws + 4 * nW + 256 + sizeof(int) * E_NUM * T_TOK);

    // 1. zero outputs + stats
    moe_init<<<1024, 256, 0, stream>>>(out, out_size, cnts, psum);

    // 2. fp32 -> bf16 conversion + WMMA-B tile swizzle
    moe_convert_pack<<<8192, 256, 0, stream>>>(W1, W1p, 5, D_DIM, H_DIM);  // K=D, N=H
    moe_convert_pack<<<8192, 256, 0, stream>>>(W2, W2p, 6, H_DIM, D_DIM);  // K=H, N=D

    // 3. gating / routing
    moe_gate<<<T_TOK, 256, 0, stream>>>(x, Wr, br, cnts, psum, tokL, wLst);

    // 4. balance loss
    moe_lbal<<<1, 32, 0, stream>>>(cnts, psum, out + (size_t)T_TOK * D_DIM);

    // 5. sparse expert MLP with bf16 WMMA
    const size_t lds_bytes = (size_t)(16 * D_DIM + 16 * H_DIM) * sizeof(ushort_t); // 96 KB
    moe_experts<<<E_NUM * (T_TOK / 16), 256, lds_bytes, stream>>>(
        x, W1p, b1, W2p, b2, cnts, tokL, wLst, out);
}
